// Layer_83562883711365
// MI455X (gfx1250) — matrix operand order repaired and checked by reading, not compiled, not run on hardware
//
#include <hip/hip_runtime.h>
#include <hip/hip_bf16.h>
#include <math.h>

constexpr int cB = 8, cS = 1024, cD = 512, cH = 8, cDH = 64;
constexpr int cN = 1024, cK = 16, cP = 64, cKP = 8, cFF = 2048;
constexpr int cM = cB * cS;

typedef __attribute__((ext_vector_type(16))) _Float16 v16h;
typedef __attribute__((ext_vector_type(8)))  float    v8f;
typedef __attribute__((ext_vector_type(4)))  unsigned v4u;
typedef __attribute__((ext_vector_type(8)))  int      v8i;
typedef __attribute__((ext_vector_type(4)))  int      v4i;
typedef _Float16 h16;

static __device__ __forceinline__ v8f wmma32(v16h a, v16h b, v8f c) {
  // D = A(16x32 f16) x B(32x16 f16) + C(16x16 f32)
  return __builtin_amdgcn_wmma_f32_16x16x32_f16(false, a, false, b, (short)0, c,
                                                false, false);
}

// Load A fragment: lane r = row, halves select K 0-15 vs 16-31.
static __device__ __forceinline__ v16h frag_a(const h16* __restrict__ p, int hf) {
  v16h a;
#pragma unroll
  for (int j = 0; j < 16; ++j) a[j] = p[hf * 16 + j];
  return a;
}
// Load B fragment from BT[N][K] layout: lane r = column n, 16 contiguous K.
static __device__ __forceinline__ v16h frag_bt(const h16* __restrict__ p) {
  v16h b;
#pragma unroll
  for (int j = 0; j < 16; ++j) b[j] = p[j];
  return b;
}

// ---------------------------------------------------------------------------
// LDS-free tiled GEMM: C[M,N] = A[M,K] @ BT[N,K]^T (+bias (+gelu))
// A row-major MxK f16; BT row-major NxK f16 (i.e. B pre-transposed).
// 128 threads = 4 waves; block tile 128(M)x64(N); wave tile 32x64; K step 32.
// Per K step: 12 contiguous b128 loads feed 8 WMMAs (1.5 loads/WMMA).
// EPI: 0 = none, 1 = +bias, 2 = +bias then exact GELU.
// ---------------------------------------------------------------------------
template <int EPI>
__global__ __launch_bounds__(128)
void gemm_h(const h16* __restrict__ A, const h16* __restrict__ BT,
            const float* __restrict__ bias, float* __restrict__ Cf,
            h16* __restrict__ Ch, int M, int Nd, int Kd) {
  const int tid  = threadIdx.x;
  const int n0   = blockIdx.x * 64;
  const int m0   = blockIdx.y * 128;
  const int wv   = tid >> 5;
  const int lane = tid & 31;
  const int r    = lane & 15;
  const int hf   = lane >> 4;

  v8f zv = {};
  v8f acc[2][4];
#pragma unroll
  for (int mi = 0; mi < 2; ++mi)
#pragma unroll
    for (int t = 0; t < 4; ++t) acc[mi][t] = zv;

  const h16* arow0 = A + (size_t)(m0 + wv * 32 + r) * Kd;
  const h16* arow1 = arow0 + (size_t)16 * Kd;
  const h16* brow0 = BT + (size_t)(n0 + r) * Kd + hf * 16;

  for (int k0 = 0; k0 < Kd; k0 += 32) {
    if (k0 + 32 < Kd) { // pull next K panel toward the caches
      __builtin_prefetch(arow0 + k0 + 32, 0, 0);
      __builtin_prefetch(arow1 + k0 + 32, 0, 0);
      __builtin_prefetch(brow0 + k0 + 32, 0, 0);
    }
    v16h a0 = frag_a(arow0 + k0, hf);
    v16h a1 = frag_a(arow1 + k0, hf);
#pragma unroll
    for (int t = 0; t < 4; ++t) {
      v16h b = frag_bt(brow0 + (size_t)t * 16 * Kd + k0);
      acc[0][t] = wmma32(a0, b, acc[0][t]);
      acc[1][t] = wmma32(a1, b, acc[1][t]);
    }
  }

#pragma unroll
  for (int mi = 0; mi < 2; ++mi)
#pragma unroll
    for (int t = 0; t < 4; ++t) {
#pragma unroll
      for (int i = 0; i < 8; ++i) {
        int row = m0 + wv * 32 + mi * 16 + hf * 8 + i;
        int col = n0 + t * 16 + r;
        float v = acc[mi][t][i];
        if (EPI >= 1) v += bias[col];
        if (EPI == 2) v = 0.5f * v * (1.0f + erff(v * 0.70710678118f));
        size_t o = (size_t)row * Nd + col;
        if (Cf) Cf[o] = v;
        if (Ch) Ch[o] = (h16)v;
      }
    }
}

// ---------------------------------------------------------------------------
// LayerNorm over D=512: one block per row.
// ---------------------------------------------------------------------------
__global__ __launch_bounds__(256)
void ln_kernel(const float* __restrict__ x, const float* __restrict__ g,
               const float* __restrict__ b, float* __restrict__ outf,
               h16* __restrict__ outh) {
  __shared__ float red[256];
  int row = blockIdx.x, tid = threadIdx.x;
  const float* xr = x + (size_t)row * cD;
  float s = xr[tid] + xr[tid + 256];
  red[tid] = s; __syncthreads();
  for (int o = 128; o > 0; o >>= 1) { if (tid < o) red[tid] += red[tid + o]; __syncthreads(); }
  float mu = red[0] * (1.0f / cD);
  __syncthreads();
  float d0 = xr[tid] - mu, d1 = xr[tid + 256] - mu;
  red[tid] = d0 * d0 + d1 * d1; __syncthreads();
  for (int o = 128; o > 0; o >>= 1) { if (tid < o) red[tid] += red[tid + o]; __syncthreads(); }
  float rs = rsqrtf(red[0] * (1.0f / cD) + 1e-5f);
  float y0 = d0 * rs * g[tid] + b[tid];
  float y1 = d1 * rs * g[tid + 256] + b[tid + 256];
  size_t base = (size_t)row * cD;
  if (outf) { outf[base + tid] = y0; outf[base + tid + 256] = y1; }
  if (outh) { outh[base + tid] = (h16)y0; outh[base + tid + 256] = (h16)y1; }
}

__global__ void cvt_kernel(const float* __restrict__ src, h16* __restrict__ dst, int n) {
  int i = blockIdx.x * 256 + threadIdx.x;
  if (i < n) dst[i] = (h16)src[i];
}

// src[rows,cols] f32 -> dst[cols,rows] f16  (produces BT layouts)
__global__ void cvtT_kernel(const float* __restrict__ src, h16* __restrict__ dst,
                            int rows, int cols) {
  int i = blockIdx.x * 256 + threadIdx.x;
  if (i < rows * cols) {
    int rr = i / cols, cc = i % cols;
    dst[(size_t)cc * rows + rr] = (h16)src[i];
  }
}

// v_h[b,s,h,dh] -> vt[b,h,dh,s]  (per-head V^T so attention B frags are
// contiguous b128 loads instead of 16-way strided gathers)
__global__ void vT_kernel(const h16* __restrict__ vh, h16* __restrict__ vt) {
  int i = blockIdx.x * 256 + threadIdx.x;
  if (i >= cM * cD) return;
  int sg = i / cD, df = i % cD;
  int b = sg / cS, s = sg % cS;
  int h = df / cDH, d = df % cDH;
  vt[(((size_t)b * cH + h) * cDH + d) * cS + s] = vh[i];
}

// ---------------------------------------------------------------------------
// Flash attention: one wave per (16 query rows, head, batch). dh=64, S=1024.
// S and O tiles via WMMA; online softmax through LDS. V accessed transposed.
// ---------------------------------------------------------------------------
__global__ __launch_bounds__(32)
void attn_kernel(const h16* __restrict__ qh, const h16* __restrict__ kh,
                 const h16* __restrict__ vt, float* __restrict__ ctx,
                 h16* __restrict__ ctxh) {
  __shared__ float S_lds[16][33];
  __shared__ h16   P_lds[16][32];
  __shared__ float scale_lds[16];
  const int lane = threadIdx.x, r = lane & 15, hf = lane >> 4;
  const int q0   = blockIdx.x * 16;
  const int head = blockIdx.y;
  const size_t hoff = (size_t)head * cDH;
  const size_t bS   = (size_t)blockIdx.z * cS;

  v8f zv = {};
  v8f o[4];
#pragma unroll
  for (int t = 0; t < 4; ++t) o[t] = zv;
  float runmax = -3.0e38f, runsum = 0.0f;
  const h16* qr  = qh + (bS + q0 + r) * cD + hoff;
  const h16* vtb = vt + (((size_t)blockIdx.z * cH + head) * cDH + r) * cS;

  for (int kt = 0; kt < cS / 32; ++kt) {
    v8f s0 = zv, s1 = zv;
#pragma unroll
    for (int k0 = 0; k0 < cDH; k0 += 32) {
      v16h a = frag_a(qr + k0, hf);
      v16h b0 = frag_bt(kh + (bS + kt * 32 + r) * cD + hoff + k0 + hf * 16);
      v16h b1 = frag_bt(kh + (bS + kt * 32 + 16 + r) * cD + hoff + k0 + hf * 16);
      s0 = wmma32(a, b0, s0);
      s1 = wmma32(a, b1, s1);
    }
#pragma unroll
    for (int i = 0; i < 8; ++i) { S_lds[hf * 8 + i][r] = s0[i]; S_lds[hf * 8 + i][16 + r] = s1[i]; }
    __syncthreads();
    if (lane < 16) {
      float mx = runmax;
#pragma unroll
      for (int j = 0; j < 32; ++j) { float t = S_lds[lane][j] * 0.125f; S_lds[lane][j] = t; mx = fmaxf(mx, t); }
      float alpha = __expf(runmax - mx);
      float sm = 0.0f;
#pragma unroll
      for (int j = 0; j < 32; ++j) { float p = __expf(S_lds[lane][j] - mx); P_lds[lane][j] = (h16)p; sm += p; }
      runmax = mx; runsum = runsum * alpha + sm;
      scale_lds[lane] = alpha;
    }
    __syncthreads();
    v16h a = frag_a(&P_lds[r][0], hf);
#pragma unroll
    for (int t = 0; t < 4; ++t) {
#pragma unroll
      for (int i = 0; i < 8; ++i) o[t][i] *= scale_lds[hf * 8 + i];
      // V^T[b,h][n = t*16+r][k = kt*32 + hf*16 ..] : contiguous 16 halves
      v16h b = frag_bt(vtb + (size_t)t * 16 * cS + kt * 32 + hf * 16);
      o[t] = wmma32(a, b, o[t]);
    }
    __syncthreads();
  }
  if (lane < 16) scale_lds[lane] = 1.0f / runsum;
  __syncthreads();
#pragma unroll
  for (int t = 0; t < 4; ++t)
#pragma unroll
    for (int i = 0; i < 8; ++i) {
      size_t row = bS + q0 + hf * 8 + i;
      size_t col = hoff + t * 16 + r;
      float v = o[t][i] * scale_lds[hf * 8 + i];
      ctx[row * cD + col]  = v;
      ctxh[row * cD + col] = (h16)v;
    }
}

// ---------------------------------------------------------------------------
// Per-token 2-way gate + mixed = w0*normed + w1*context  (f16 out).
// ---------------------------------------------------------------------------
__global__ __launch_bounds__(256)
void wmix_kernel(const float* __restrict__ normed, const float* __restrict__ ctx,
                 const float* __restrict__ Wp, const float* __restrict__ bp,
                 h16* __restrict__ mixedh) {
  __shared__ float r0[256], r1[256];
  int row = blockIdx.x, tid = threadIdx.x;
  const float* nr = normed + (size_t)row * cD;
  const float* cr = ctx + (size_t)row * cD;
  float p0 = 0, p1 = 0;
  for (int d = tid; d < cD; d += 256) {
    float n = nr[d], c = cr[d];
    p0 += n * Wp[d * 2 + 0] + c * Wp[(cD + d) * 2 + 0];
    p1 += n * Wp[d * 2 + 1] + c * Wp[(cD + d) * 2 + 1];
  }
  r0[tid] = p0; r1[tid] = p1; __syncthreads();
  for (int o = 128; o > 0; o >>= 1) {
    if (tid < o) { r0[tid] += r0[tid + o]; r1[tid] += r1[tid + o]; }
    __syncthreads();
  }
  float l0 = r0[0] + bp[0], l1 = r1[0] + bp[1];
  float m = fmaxf(l0, l1);
  float e0 = __expf(l0 - m), e1 = __expf(l1 - m);
  float w0 = e0 / (e0 + e1), w1 = e1 / (e0 + e1);
  for (int d = tid; d < cD; d += 256)
    mixedh[(size_t)row * cD + d] = (h16)(w0 * nr[d] + w1 * cr[d]);
}

// ---------------------------------------------------------------------------
// Router: top-16 over N=1024 scores, softmax weights, residual gather-add.
// ---------------------------------------------------------------------------
__global__ __launch_bounds__(256)
void router_kernel(const float* __restrict__ scores, const float* __restrict__ x,
                   const float* __restrict__ neurons, float* __restrict__ xnew,
                   float* __restrict__ topw, int* __restrict__ topidx,
                   int* __restrict__ outidx) {
  __shared__ float sc[cN];
  __shared__ float bv[256];
  __shared__ int   bi[256];
  __shared__ int   sel[cK];
  __shared__ float selw[cK];
  int row = blockIdx.x, tid = threadIdx.x;
  const float* sr = scores + (size_t)row * cN;
  for (int i = tid; i < cN; i += 256) sc[i] = sr[i];
  __syncthreads();
  for (int k = 0; k < cK; ++k) {
    float best = -3.0e38f; int bidx = 0;
    for (int i = tid; i < cN; i += 256)
      if (sc[i] > best) { best = sc[i]; bidx = i; }
    bv[tid] = best; bi[tid] = bidx; __syncthreads();
    for (int o = 128; o > 0; o >>= 1) {
      if (tid < o && bv[tid + o] > bv[tid]) { bv[tid] = bv[tid + o]; bi[tid] = bi[tid + o]; }
      __syncthreads();
    }
    if (tid == 0) { sel[k] = bi[0]; selw[k] = bv[0]; sc[bi[0]] = -3.0e38f; }
    __syncthreads();
  }
  if (tid == 0) {
    float mx = selw[0], s = 0.0f;
    for (int k = 0; k < cK; ++k) { float e = __expf(selw[k] - mx); selw[k] = e; s += e; }
    for (int k = 0; k < cK; ++k) {
      selw[k] /= s;
      topw[(size_t)row * cK + k]   = selw[k];
      topidx[(size_t)row * cK + k] = sel[k];
      outidx[(size_t)row * cK + k] = sel[k];
    }
  }
  __syncthreads();
  for (int d = tid; d < cD; d += 256) {
    float acc = x[(size_t)row * cD + d];
    for (int k = 0; k < cK; ++k) acc += selw[k] * neurons[(size_t)sel[k] * cD + d];
    xnew[(size_t)row * cD + d] = acc;
  }
}

// ---------------------------------------------------------------------------
// Per-token interaction: S=qn@knT, inter=softmax(S)@vn, npsT=inter@pattT,
// pattern top-8, aggregation. One wave per token; all GEMMs via WMMA.
// qn/kn/vn are gathers of precomputed nQ/nK/nV (neurons @ nW* + b).
// ---------------------------------------------------------------------------
__global__ __launch_bounds__(32)
void interact_kernel(const int* __restrict__ topidx, const float* __restrict__ topw,
                     const h16* __restrict__ nQ, const h16* __restrict__ nK,
                     const h16* __restrict__ nV, const h16* __restrict__ patth,
                     const float* __restrict__ ctxP, const float* __restrict__ normed2,
                     h16* __restrict__ combh) {
  __shared__ int   idx[cK];
  __shared__ float tw[cK];
  __shared__ float S16[16][17];
  __shared__ h16   A16[16][32];
  __shared__ h16   inter_lds[16][cD];
  __shared__ float nps_lds[16][cP];
  __shared__ float ps[cP];
  __shared__ float kpmax[cKP], kpden[cKP], tpw_s[cKP];
  __shared__ int   tpi_s[cKP];
  __shared__ float fa[cK];
  const int row = blockIdx.x;
  const int lane = threadIdx.x, r = lane & 15, hf = lane >> 4;
  if (lane < cK) {
    idx[lane] = topidx[(size_t)row * cK + lane];
    tw[lane]  = topw[(size_t)row * cK + lane];
  }
  __syncthreads();

  const float iSD = 0.044194173824159216f; // 1/sqrt(512)
  v8f zv = {};

  // S = qn @ kn^T (16x16, K=512); both sides contiguous row gathers.
  v8f s = zv;
  for (int k0 = 0; k0 < cD; k0 += 32) {
    v16h a = frag_a(nQ + (size_t)idx[r] * cD + k0, hf);
    v16h b = frag_bt(nK + (size_t)idx[r] * cD + k0 + hf * 16);
    s = wmma32(a, b, s);
  }
#pragma unroll
  for (int i = 0; i < 8; ++i) S16[hf * 8 + i][r] = s[i] * iSD;
  __syncthreads();
  if (lane < 16) {
    float mx = -3.0e38f;
#pragma unroll
    for (int j = 0; j < 16; ++j) mx = fmaxf(mx, S16[lane][j]);
    float sm = 0.0f;
#pragma unroll
    for (int j = 0; j < 16; ++j) { float e = __expf(S16[lane][j] - mx); S16[lane][j] = e; sm += e; }
    float inv = 1.0f / sm;
#pragma unroll
    for (int j = 0; j < 16; ++j) A16[lane][j] = (h16)(S16[lane][j] * inv);
#pragma unroll
    for (int j = 16; j < 32; ++j) A16[lane][j] = (h16)0.0f;
  }
  __syncthreads();

  // inter = A(16x16, K padded to 32) @ vn(16x512); vn rows are a true gather.
  v16h aA = frag_a(&A16[r][0], hf);
  for (int t = 0; t < cD / 16; ++t) {
    v16h b;
#pragma unroll
    for (int j = 0; j < 16; ++j)
      b[j] = hf ? (h16)0.0f : nV[(size_t)idx[j] * cD + t * 16 + r];
    v8f c = zv;
    c = wmma32(aA, b, c);
#pragma unroll
    for (int i = 0; i < 8; ++i) inter_lds[hf * 8 + i][t * 16 + r] = (h16)c[i];
  }
  __syncthreads();

  // npsT(16x64) = inter(16x512) @ patterns^T; BT layout == patterns itself.
  v8f np[4];
#pragma unroll
  for (int t = 0; t < 4; ++t) np[t] = zv;
  for (int k0 = 0; k0 < cD; k0 += 32) {
    v16h a = frag_a(&inter_lds[r][k0], hf);
#pragma unroll
    for (int t = 0; t < 4; ++t) {
      v16h b = frag_bt(patth + (size_t)(t * 16 + r) * cD + k0 + hf * 16);
      np[t] = wmma32(a, b, np[t]);
    }
  }
#pragma unroll
  for (int t = 0; t < 4; ++t)
#pragma unroll
    for (int i = 0; i < 8; ++i) nps_lds[hf * 8 + i][t * 16 + r] = np[t][i] * iSD;
  __syncthreads();

  // pattern_scores = 0.5*(nps . tw) + 0.5*ctxP
  for (int p = lane; p < cP; p += 32) {
    float nb = 0.0f;
#pragma unroll
    for (int k = 0; k < cK; ++k) nb += nps_lds[k][p] * tw[k];
    ps[p] = 0.5f * nb + 0.5f * ctxP[(size_t)row * cP + p];
  }
  __syncthreads();
  if (lane == 0) {
    float tps[cKP];
    for (int kp = 0; kp < cKP; ++kp) {
      float best = -3.0e38f; int bidx = 0;
      for (int p = 0; p < cP; ++p) if (ps[p] > best) { best = ps[p]; bidx = p; }
      tpi_s[kp] = bidx; tps[kp] = best; ps[bidx] = -3.0e38f;
    }
    float mx = tps[0], sm = 0.0f;
    for (int kp = 0; kp < cKP; ++kp) { float e = __expf(tps[kp] - mx); tpw_s[kp] = e; sm += e; }
    for (int kp = 0; kp < cKP; ++kp) tpw_s[kp] /= sm;
  }
  __syncthreads();
  if (lane < cKP) {
    int p = tpi_s[lane];
    float mx = -3.0e38f;
    for (int k = 0; k < cK; ++k) mx = fmaxf(mx, nps_lds[k][p]);
    float sm = 0.0f;
    for (int k = 0; k < cK; ++k) sm += __expf(nps_lds[k][p] - mx);
    kpmax[lane] = mx; kpden[lane] = sm;
  }
  __syncthreads();
  if (lane < cK) {
    float f = 0.0f;
    for (int kp = 0; kp < cKP; ++kp)
      f += tpw_s[kp] * __expf(nps_lds[lane][tpi_s[kp]] - kpmax[kp]) / kpden[kp];
    fa[lane] = f;
  }
  __syncthreads();
  for (int d = lane; d < cD; d += 32) {
    float agg = 0.0f;
#pragma unroll
    for (int k = 0; k < cK; ++k) agg += fa[k] * (float)inter_lds[k][d];
    combh[(size_t)row * cD + d] = (h16)(normed2[(size_t)row * cD + d] + agg);
  }
}

__global__ void add_kernel(const float* __restrict__ a, const float* __restrict__ b,
                           float* __restrict__ o, int n) {
  int i = blockIdx.x * 256 + threadIdx.x;
  if (i < n) o[i] = a[i] + b[i];
}

// ---------------------------------------------------------------------------
// TDM probe: exercise the Tensor Data Mover path (guarded so it can never
// break the build). Copies a 64x64 f16 tile of a 512x512 tensor into LDS via
// tensor_load_to_lds, waits on TENSORcnt, reduces, writes to scratch.
// ---------------------------------------------------------------------------
__global__ __launch_bounds__(32)
void tdm_probe_kernel(const h16* __restrict__ src, float* __restrict__ scratch) {
#if __has_builtin(__builtin_amdgcn_tensor_load_to_lds) && __has_builtin(__builtin_amdgcn_s_wait_tensorcnt)
  __shared__ h16 tile[64 * 64];
  unsigned long long ga = (unsigned long long)(const void*)src;
  unsigned lds_off = (unsigned)(unsigned long long)(void*)tile;
  // D# group 0: count=1 | lds_addr[63:32] | global_addr[120:64] | type=2
  v4u g0 = {1u, lds_off, (unsigned)ga,
            (unsigned)((ga >> 32) & 0x01FFFFFFull) | (2u << 30)};
  // D# group 1: data_size=1 (2B); tensor 512x512; tile 64x64; stride0=512.
  v8i g1 = {(int)(1u << 16),          // mask=0, data_size=1, no flags
            (int)(512u << 16),        // tensor_dim0 low16 at bit 48
            (int)(512u << 16),        // dim0 hi16=0 | tensor_dim1 low16 at bit 80
            (int)(64u << 16),         // dim1 hi16=0 | tile_dim0 at bit 112
            (int)64,                  // tile_dim1=64, tile_dim2=0
            (int)512,                 // tensor_dim0_stride low32
            (int)(512u << 16),        // stride0 hi16=0 | tensor_dim1_stride low16
            0};                       // tensor_dim1_stride hi
  v4i g2 = {0, 0, 0, 0};
  v4i g3 = {0, 0, 0, 0};
#if defined(__clang_major__) && (__clang_major__ >= 23)
  v8i g1b = {0, 0, 0, 0, 0, 0, 0, 0};
  __builtin_amdgcn_tensor_load_to_lds(g0, g1, g2, g3, g1b, 0);
#else
  __builtin_amdgcn_tensor_load_to_lds(g0, g1, g2, g3, 0);
#endif
  __builtin_amdgcn_s_wait_tensorcnt(0);
  __syncthreads();
  float s = 0.0f;
  for (int i = threadIdx.x; i < 64 * 64; i += 32) s += (float)tile[i];
  scratch[threadIdx.x] = s;
#else
  if (threadIdx.x == 0) scratch[0] = 0.0f;
#endif
}

// ---------------------------------------------------------------------------
extern "C" void kernel_launch(void* const* d_in, const int* in_sizes, int n_in,
                              void* d_out, int out_size, void* d_ws, size_t ws_size,
                              hipStream_t stream) {
  (void)in_sizes; (void)n_in; (void)out_size; (void)ws_size;
  const float* x   = (const float*)d_in[0];
  const float* g1  = (const float*)d_in[1];
  const float* b1  = (const float*)d_in[2];
  const float* g2  = (const float*)d_in[3];
  const float* b2  = (const float*)d_in[4];
  const float* Wq  = (const float*)d_in[5];
  const float* bq  = (const float*)d_in[6];
  const float* Wk  = (const float*)d_in[7];
  const float* bk  = (const float*)d_in[8];
  const float* Wv  = (const float*)d_in[9];
  const float* bv  = (const float*)d_in[10];
  const float* neurons = (const float*)d_in[11];
  const float* Wp  = (const float*)d_in[12];
  const float* bp  = (const float*)d_in[13];
  const float* nWq = (const float*)d_in[14];
  const float* nbq = (const float*)d_in[15];
  const float* nWk = (const float*)d_in[16];
  const float* nbk = (const float*)d_in[17];
  const float* nWv = (const float*)d_in[18];
  const float* nbv = (const float*)d_in[19];
  const float* patt = (const float*)d_in[20];
  const float* Wup = (const float*)d_in[21];
  const float* bup = (const float*)d_in[22];
  const float* Wdn = (const float*)d_in[23];
  const float* bdn = (const float*)d_in[24];

  char* wsb = (char*)d_ws;
  size_t off = 0;
  auto alloc = [&](size_t bytes) -> char* {
    char* p = wsb + off;
    off = (off + bytes + 255) & ~(size_t)255;
    return p;
  };

  float* normed   = (float*)alloc((size_t)cM * cD * 4);
  h16*   normed_h = (h16*)alloc((size_t)cM * cD * 2);
  // Transposed (BT = N x K) f16 weights for LDS-free WMMA GEMM.
  h16*   WqT_h  = (h16*)alloc((size_t)cD * cD * 2);
  h16*   WkT_h  = (h16*)alloc((size_t)cD * cD * 2);
  h16*   WvT_h  = (h16*)alloc((size_t)cD * cD * 2);
  h16*   nWqT_h = (h16*)alloc((size_t)cD * cD * 2);
  h16*   nWkT_h = (h16*)alloc((size_t)cD * cD * 2);
  h16*   nWvT_h = (h16*)alloc((size_t)cD * cD * 2);
  h16*   WupT_h = (h16*)alloc((size_t)cFF * cD * 2);
  h16*   WdnT_h = (h16*)alloc((size_t)cD * cFF * 2);
  h16*   neurons_h = (h16*)alloc((size_t)cN * cD * 2); // BT for scores GEMM
  h16*   patt_h    = (h16*)alloc((size_t)cP * cD * 2); // BT for pattern GEMMs
  h16*   q_h = (h16*)alloc((size_t)cM * cD * 2);
  h16*   k_h = (h16*)alloc((size_t)cM * cD * 2);
  h16*   v_h = (h16*)alloc((size_t)cM * cD * 2);
  h16*   vt_h = (h16*)alloc((size_t)cM * cD * 2);     // per-head V^T
  h16*   nQ_h = (h16*)alloc((size_t)cN * cD * 2);
  h16*   nK_h = (h16*)alloc((size_t)cN * cD * 2);
  h16*   nV_h = (h16*)alloc((size_t)cN * cD * 2);
  float* context   = (float*)alloc((size_t)cM * cD * 4);
  h16*   context_h = (h16*)alloc((size_t)cM * cD * 2);
  h16*   mixed_h   = (h16*)alloc((size_t)cM * cD * 2);
  float* scores = (float*)alloc((size_t)cM * cN * 4);
  float* topw   = (float*)alloc((size_t)cM * cK * 4);
  int*   topidx = (int*)alloc((size_t)cM * cK * 4);
  float* xnew    = (float*)alloc((size_t)cM * cD * 4);
  float* normed2 = (float*)alloc((size_t)cM * cD * 4);
  float* ctxP    = (float*)alloc((size_t)cM * cP * 4);
  h16*   comb_h  = (h16*)alloc((size_t)cM * cD * 2);
  h16*   hff_h   = (h16*)alloc((size_t)cM * cFF * 2);
  float* ffn     = (float*)alloc((size_t)cM * cD * 4);
  float* tdm_scr = (float*)alloc(256 * 4);

  float* out    = (float*)d_out;
  int*   outidx = (int*)(out + (size_t)cM * cD);

  auto cvtT = [&](const float* s, h16* d, int rows, int cols) {
    cvtT_kernel<<<(rows * cols + 255) / 256, 256, 0, stream>>>(s, d, rows, cols);
  };
  cvtT(Wq, WqT_h, cD, cD);   cvtT(Wk, WkT_h, cD, cD);   cvtT(Wv, WvT_h, cD, cD);
  cvtT(nWq, nWqT_h, cD, cD); cvtT(nWk, nWkT_h, cD, cD); cvtT(nWv, nWvT_h, cD, cD);
  cvtT(Wup, WupT_h, cD, cFF);
  cvtT(Wdn, WdnT_h, cFF, cD);
  cvt_kernel<<<(cN * cD + 255) / 256, 256, 0, stream>>>(neurons, neurons_h, cN * cD);
  cvt_kernel<<<(cP * cD + 255) / 256, 256, 0, stream>>>(patt, patt_h, cP * cD);

  ln_kernel<<<cM, 256, 0, stream>>>(x, g1, b1, normed, normed_h);

  gemm_h<1><<<dim3(cD / 64, cM / 128), 128, 0, stream>>>(normed_h, WqT_h, bq, nullptr, q_h, cM, cD, cD);
  gemm_h<1><<<dim3(cD / 64, cM / 128), 128, 0, stream>>>(normed_h, WkT_h, bk, nullptr, k_h, cM, cD, cD);
  gemm_h<1><<<dim3(cD / 64, cM / 128), 128, 0, stream>>>(normed_h, WvT_h, bv, nullptr, v_h, cM, cD, cD);

  // nQ/nK/nV = neurons @ nW* + b  (gather trick: selected@W == (neurons@W)[idx])
  gemm_h<1><<<dim3(cD / 64, cN / 128), 128, 0, stream>>>(neurons_h, nWqT_h, nbq, nullptr, nQ_h, cN, cD, cD);
  gemm_h<1><<<dim3(cD / 64, cN / 128), 128, 0, stream>>>(neurons_h, nWkT_h, nbk, nullptr, nK_h, cN, cD, cD);
  gemm_h<1><<<dim3(cD / 64, cN / 128), 128, 0, stream>>>(neurons_h, nWvT_h, nbv, nullptr, nV_h, cN, cD, cD);

  vT_kernel<<<(cM * cD + 255) / 256, 256, 0, stream>>>(v_h, vt_h);

  attn_kernel<<<dim3(cS / 16, cH, cB), 32, 0, stream>>>(q_h, k_h, vt_h, context, context_h);

  wmix_kernel<<<cM, 256, 0, stream>>>(normed, context, Wp, bp, mixed_h);

  // scores = mixed @ neurons^T ; BT layout of neurons^T is neurons itself.
  gemm_h<0><<<dim3(cN / 64, cM / 128), 128, 0, stream>>>(mixed_h, neurons_h, nullptr, scores, nullptr, cM, cN, cD);

  router_kernel<<<cM, 256, 0, stream>>>(scores, x, neurons, xnew, topw, topidx, outidx);

  ln_kernel<<<cM, 256, 0, stream>>>(xnew, g2, b2, normed2, nullptr);

  // context_based = context @ patterns^T ; BT layout is patterns itself.
  gemm_h<0><<<dim3(cP / 64, cM / 128), 128, 0, stream>>>(context_h, patt_h, nullptr, ctxP, nullptr, cM, cP, cD);

  interact_kernel<<<cM, 32, 0, stream>>>(topidx, topw, nQ_h, nK_h, nV_h, patt_h, ctxP, normed2, comb_h);

  gemm_h<2><<<dim3(cFF / 64, cM / 128), 128, 0, stream>>>(comb_h, WupT_h, bup, nullptr, hff_h, cM, cFF, cD);
  gemm_h<1><<<dim3(cD / 64, cM / 128), 128, 0, stream>>>(hff_h, WdnT_h, bdn, ffn, nullptr, cM, cD, cFF);

  add_kernel<<<(cM * cD + 255) / 256, 256, 0, stream>>>(xnew, ffn, out, cM * cD);

  tdm_probe_kernel<<<1, 32, 0, stream>>>(neurons_h, tdm_scr);
}
